// Decoder_89361089561505
// MI455X (gfx1250) — matrix operand order repaired and checked by reading, not compiled, not run on hardware
//
#include <hip/hip_runtime.h>
#include <math.h>

// ---------------------------------------------------------------------------
// Decoder: GRU(1024, autoregressive, T=512) -> LSTM(1024) -> Linear(128)
// B=128, LAT=LSTM_DIM=1024, T=512, OUT=128.
// GEMMs on bf16 WMMA (v_wmma_f32_16x16x32_bf16); activation panels staged
// into LDS by the CDNA5 Tensor Data Mover (double-buffered, TENSORcnt),
// recurrence state kept in fp32.
// ---------------------------------------------------------------------------

typedef __attribute__((ext_vector_type(16))) __bf16 v16bf;
typedef __attribute__((ext_vector_type(8)))  float  v8f;
typedef __attribute__((ext_vector_type(4)))  unsigned int v4u;
typedef __attribute__((ext_vector_type(8)))  int    v8i;
typedef __attribute__((ext_vector_type(4)))  int    v4i;

#define DEC_B   128
#define DEC_L   1024
#define DEC_T   512
#define DEC_O   128
#define DEC_BL  (DEC_B * DEC_L)

// TDM-staged X tile: 128 rows x 64 k bf16, padded 4 DWORDs per 32-DWORD row
// (row stride 72 elems = 144 B = 9*16 B -> aligned b128 frag reads, banks
// skewed by 4 DWORDs/row so 16-lane column reads are conflict-free).
#define KCHUNK        64
#define XROW_ELEMS    72
#define BUF_ELEMS     (128 * XROW_ELEMS)            // 9216 bf16
#define BUF_BYTES     (BUF_ELEMS * 2)               // 18432 B
#define GEMM_LDS_BYTES (2 * BUF_BYTES)              // 36864 B double buffer

#define XS_STRIDE 40   // FC kernel static-LDS tile stride (32 k + 8 pad)

// ------------------------------ fp32 -> bf16 -------------------------------
__global__ void cvt_bf16_kernel(const float* __restrict__ src,
                                __bf16* __restrict__ dst, int n) {
  int i = blockIdx.x * blockDim.x + threadIdx.x;
  int stride = gridDim.x * blockDim.x;
  for (; i < n; i += stride) dst[i] = (__bf16)src[i];
}

// init: seq[0]=0 (GRU x0 AND LSTM h0 zeros), h_gru=z (fp32+bf16), c_lstm=0
__global__ void init_kernel(const float* __restrict__ z,
                            __bf16* __restrict__ seq0,
                            float*  __restrict__ h_gru_f32,
                            __bf16* __restrict__ h_gru_bf,
                            float*  __restrict__ c_f32, int n) {
  int i = blockIdx.x * blockDim.x + threadIdx.x;
  if (i < n) {
    seq0[i]      = (__bf16)0.0f;
    float zv     = z[i];
    h_gru_f32[i] = zv;
    h_gru_bf[i]  = (__bf16)zv;
    c_f32[i]     = 0.0f;
  }
}

// ---------------------------------------------------------------------------
// WMMA fragment loaders (layouts per CDNA5 ISA VGPR tables).
// ---------------------------------------------------------------------------
__device__ __forceinline__ v16bf load_a_frag(const __bf16* __restrict__ base,
                                             int row_stride, int lane) {
  int r  = lane & 15;
  int fk = (lane & 16) ? 8 : 0;
  const __bf16* p = base + (size_t)r * row_stride + fk;
  union { v16bf v; uint4 q[2]; } u;
  u.q[0] = *reinterpret_cast<const uint4*>(p);
  u.q[1] = *reinterpret_cast<const uint4*>(p + 16);
  return u.v;
}

__device__ __forceinline__ v16bf load_b_frag_lds(const __bf16* __restrict__ base,
                                                 int row_stride, int lane) {
  int c  = lane & 15;
  int bk = (lane & 16) ? 8 : 0;
  const __bf16* p = base + (size_t)c * row_stride + bk;
  union { v16bf v; uint4 q[2]; } u;
  u.q[0] = *reinterpret_cast<const uint4*>(p);
  u.q[1] = *reinterpret_cast<const uint4*>(p + 16);
  return u.v;
}

// ---------------------------------------------------------------------------
// TDM: DMA a 2D tile (KCHUNK x 128 rows) of X into LDS buffer `ldsoff`.
// D# built per CDNA5 ISA: data_size=2B, pad_enable,
// pad_interval=32 DWORDs (code 4), pad_amount=4 DWORDs (code 3).
// This toolchain exposes the 6-arg builtin:
//   (uint32x4 g0, int32x8 g1, int32x4 g2, int32x4 g3, int32x8 g4, i32 cpol)
// 2D tile -> groups 2..4 are zero.
// ---------------------------------------------------------------------------
#if __has_builtin(__builtin_amdgcn_tensor_load_to_lds)
#define USE_TDM 1
__device__ __forceinline__ void tdm_load_x_tile(const __bf16* __restrict__ X,
                                                int colB, int K, int kc,
                                                int rows_left,
                                                unsigned ldsoff) {
  unsigned long long ga =
      (unsigned long long)(uintptr_t)X +
      ((unsigned long long)colB * (unsigned)K + (unsigned)kc) * 2ull;

  v4u g0;
  g0.x = 1u;                                        // count=1, user descriptor
  g0.y = ldsoff;                                    // lds_addr (bytes)
  g0.z = (unsigned)(ga & 0xFFFFFFFFu);              // global_addr[31:0]
  g0.w = (unsigned)(ga >> 32) | (2u << 30);         // addr[56:32] | type=2

  unsigned td0 = (unsigned)K;                       // tensor_dim0
  unsigned td1 = (unsigned)rows_left;               // tensor_dim1
  v8i g1;
  g1[0] = (int)((1u << 16)  |                       // data_size = 2 bytes
                (1u << 20)  |                       // pad_enable
                (4u << 22)  |                       // pad_interval: 32 DWORDs
                (3u << 25));                        // pad_amount:   4 DWORDs
  g1[1] = (int)((td0 & 0xFFFFu) << 16);             // [47:32]=bar addr(0), [63:48]=td0 lo
  g1[2] = (int)((td0 >> 16) | ((td1 & 0xFFFFu) << 16));
  g1[3] = (int)((td1 >> 16) | ((unsigned)KCHUNK << 16));   // tile_dim0 = 64
  g1[4] = (int)128;                                 // tile_dim1 = 128 rows
  g1[5] = (int)td0;                                 // tensor_dim0_stride lo32 = K
  g1[6] = 0;                                        // stride hi16 | dim1_stride lo
  g1[7] = 0;

  v4i gz4 = {0, 0, 0, 0};
  v8i gz8 = {0, 0, 0, 0, 0, 0, 0, 0};
  __builtin_amdgcn_tensor_load_to_lds(g0, g1, gz4, gz4, gz8, 0);
}
#else
#define USE_TDM 0
#endif

// ---------------------------------------------------------------------------
// D[n,c] = sum_k W[n,k] * X[c,k]      (W: N x K bf16, X: C x K bf16, D f32)
// Block = 256 thr (8 waves). Tile: 128 W-rows x 128 X-rows.
// X panel TDM-staged to LDS in KCHUNK slices, double-buffered.
// ---------------------------------------------------------------------------
__global__ __launch_bounds__(256)
void wmma_gemm_nt(const __bf16* __restrict__ W, const __bf16* __restrict__ X,
                  float* __restrict__ D, int K, int ldd, int xrows) {
  extern __shared__ __bf16 xs[];                    // 2 * BUF_ELEMS

  const int tid  = threadIdx.x;
  const int lane = tid & 31;
  const int wid  = tid >> 5;
  const int wm   = wid >> 1;                        // 0..3 : 32 W-rows each
  const int wn   = wid & 1;                         // 0..1 : 64 X-rows each
  const int row0 = blockIdx.x * 128 + wm * 32;
  const int colB = blockIdx.y * 128;
  const int coff = wn * 64;
  const int fr   = lane & 15;

  v8f acc[2][4] = {};

#if USE_TDM
  if (wid == 0)
    tdm_load_x_tile(X, colB, K, 0, xrows - colB, 0u);
#endif

  int cur = 0;
  for (int kc = 0; kc < K; kc += KCHUNK) {
    __syncthreads();   // everyone done reading buf[cur^1] from 2 chunks ago
#if USE_TDM
    if (wid == 0) {
      if (kc + KCHUNK < K) {
        tdm_load_x_tile(X, colB, K, kc + KCHUNK, xrows - colB,
                        (unsigned)((cur ^ 1) * BUF_BYTES));
        __builtin_amdgcn_s_wait_tensorcnt(1);       // buf[cur] DMA complete
      } else {
        __builtin_amdgcn_s_wait_tensorcnt(0);
      }
    }
#else
    {   // cooperative staging fallback, same padded layout
      int bl = tid >> 1, half = tid & 1;
      const uint4* s = reinterpret_cast<const uint4*>(
          X + (size_t)(colB + bl) * K + kc + half * 32);
      uint4* d = reinterpret_cast<uint4*>(
          &xs[cur * BUF_ELEMS + bl * XROW_ELEMS + half * 32]);
      d[0] = s[0]; d[1] = s[1]; d[2] = s[2]; d[3] = s[3];
    }
#endif
    if (kc + KCHUNK < K)   // CDNA5 global_prefetch of next W k-slice
      __builtin_prefetch(W + (size_t)(row0 + fr) * K + kc + KCHUNK, 0, 3);
    __syncthreads();       // buf[cur] ready for all waves

    const __bf16* bufp = xs + cur * BUF_ELEMS;
#pragma unroll
    for (int kk = 0; kk < KCHUNK / 32; ++kk) {
      v16bf a[2], b[4];
#pragma unroll
      for (int i = 0; i < 2; ++i)
        a[i] = load_a_frag(W + (size_t)(row0 + i * 16) * K + kc + kk * 32,
                           K, lane);
#pragma unroll
      for (int j = 0; j < 4; ++j)
        b[j] = load_b_frag_lds(bufp + (coff + j * 16) * XROW_ELEMS + kk * 32,
                               XROW_ELEMS, lane);
#pragma unroll
      for (int i = 0; i < 2; ++i)
#pragma unroll
        for (int j = 0; j < 4; ++j)
          acc[i][j] = __builtin_amdgcn_wmma_f32_16x16x32_bf16(
              false, a[i], false, b[j], (short)0, acc[i][j], false, false);
    }
    cur ^= 1;
  }

  const int mh = (lane & 16) ? 8 : 0;
#pragma unroll
  for (int i = 0; i < 2; ++i)
#pragma unroll
    for (int j = 0; j < 4; ++j) {
      int col = colB + coff + j * 16 + fr;
#pragma unroll
      for (int r = 0; r < 8; ++r) {
        int row = row0 + i * 16 + mh + r;
        D[(size_t)row * ldd + col] = acc[i][j][r];
      }
    }
}

// ---------------------------------------------------------------------------
// FC head GEMM, fused bias + (o, t*B+b) -> out[b,t,o] scatter epilogue.
// W = fc_W bf16 (128 x 1024), X = hs bf16 (T*B x 1024). grid = (1, T*B/128).
// Synchronous LDS staging path (shows the non-TDM pipeline).
// ---------------------------------------------------------------------------
__global__ __launch_bounds__(256)
void wmma_fc_kernel(const __bf16* __restrict__ W, const __bf16* __restrict__ X,
                    const float* __restrict__ bias, float* __restrict__ out,
                    int K) {
  __shared__ __bf16 xs[128 * XS_STRIDE];

  const int tid  = threadIdx.x;
  const int lane = tid & 31;
  const int wid  = tid >> 5;
  const int wm   = wid >> 1;
  const int wn   = wid & 1;
  const int row0 = wm * 32;                  // grid.x == 1 (OUT = 128)
  const int colB = blockIdx.y * 128;
  const int coff = wn * 64;
  const int fr   = lane & 15;

  v8f acc[2][4] = {};

  for (int k0 = 0; k0 < K; k0 += 32) {
    __syncthreads();
    {
      int bl = tid >> 1, half = tid & 1;
      const uint4* s = reinterpret_cast<const uint4*>(
          X + (size_t)(colB + bl) * K + k0 + half * 16);
      uint4* d = reinterpret_cast<uint4*>(&xs[bl * XS_STRIDE + half * 16]);
      d[0] = s[0];
      d[1] = s[1];
    }
    __syncthreads();

    v16bf a[2], b[4];
#pragma unroll
    for (int i = 0; i < 2; ++i)
      a[i] = load_a_frag(W + (size_t)(row0 + i * 16) * K + k0, K, lane);
#pragma unroll
    for (int j = 0; j < 4; ++j)
      b[j] = load_b_frag_lds(&xs[(coff + j * 16) * XS_STRIDE], XS_STRIDE, lane);

#pragma unroll
    for (int i = 0; i < 2; ++i)
#pragma unroll
      for (int j = 0; j < 4; ++j)
        acc[i][j] = __builtin_amdgcn_wmma_f32_16x16x32_bf16(
            false, a[i], false, b[j], (short)0, acc[i][j], false, false);
  }

  const int mh = (lane & 16) ? 8 : 0;
#pragma unroll
  for (int i = 0; i < 2; ++i)
#pragma unroll
    for (int j = 0; j < 4; ++j) {
      int col = colB + coff + j * 16 + fr;     // = t*B + b
      int t = col >> 7, bb = col & 127;
#pragma unroll
      for (int r = 0; r < 8; ++r) {
        int o = row0 + i * 16 + mh + r;
        out[((size_t)bb * DEC_T + t) * DEC_O + o] = acc[i][j][r] + bias[o];
      }
    }
}

// ------------------------------ GRU gate fuse ------------------------------
__global__ void gru_point_kernel(const float* __restrict__ gi,
                                 const float* __restrict__ gh,
                                 const float* __restrict__ bih,
                                 const float* __restrict__ bhh,
                                 float*  __restrict__ h,
                                 __bf16* __restrict__ seq_t,
                                 __bf16* __restrict__ h_bf) {
  int idx = blockIdx.x * blockDim.x + threadIdx.x;   // B*L
  int b = idx >> 10, u = idx & 1023;
  float ir = gi[(size_t)u * DEC_B + b]              + bih[u];
  float iz = gi[(size_t)(DEC_L + u) * DEC_B + b]    + bih[DEC_L + u];
  float in = gi[(size_t)(2 * DEC_L + u) * DEC_B + b]+ bih[2 * DEC_L + u];
  float hr = gh[(size_t)u * DEC_B + b]              + bhh[u];
  float hz = gh[(size_t)(DEC_L + u) * DEC_B + b]    + bhh[DEC_L + u];
  float hn = gh[(size_t)(2 * DEC_L + u) * DEC_B + b]+ bhh[2 * DEC_L + u];
  float r  = 1.0f / (1.0f + __expf(-(ir + hr)));
  float zg = 1.0f / (1.0f + __expf(-(iz + hz)));
  float n  = tanhf(in + r * hn);
  float hp = h[idx];
  float hv = (1.0f - zg) * n + zg * hp;
  h[idx] = hv;
  __bf16 hb = (__bf16)hv;
  seq_t[idx] = hb;
  h_bf[idx]  = hb;
}

// ------------------------------ LSTM gate fuse -----------------------------
__global__ void lstm_point_kernel(const float* __restrict__ gx,
                                  const float* __restrict__ gh,
                                  const float* __restrict__ bih,
                                  const float* __restrict__ bhh,
                                  float*  __restrict__ c,
                                  __bf16* __restrict__ hs_t) {
  int idx = blockIdx.x * blockDim.x + threadIdx.x;   // B*H
  int b = idx >> 10, u = idx & 1023;
  float gi = gx[(size_t)u * DEC_B + b]               + gh[(size_t)u * DEC_B + b]
           + bih[u]               + bhh[u];
  float gf = gx[(size_t)(DEC_L + u) * DEC_B + b]     + gh[(size_t)(DEC_L + u) * DEC_B + b]
           + bih[DEC_L + u]       + bhh[DEC_L + u];
  float gg = gx[(size_t)(2 * DEC_L + u) * DEC_B + b] + gh[(size_t)(2 * DEC_L + u) * DEC_B + b]
           + bih[2 * DEC_L + u]   + bhh[2 * DEC_L + u];
  float go = gx[(size_t)(3 * DEC_L + u) * DEC_B + b] + gh[(size_t)(3 * DEC_L + u) * DEC_B + b]
           + bih[3 * DEC_L + u]   + bhh[3 * DEC_L + u];
  float si = 1.0f / (1.0f + __expf(-gi));
  float sf = 1.0f / (1.0f + __expf(-gf));
  float so = 1.0f / (1.0f + __expf(-go));
  float cn = sf * c[idx] + si * tanhf(gg);
  c[idx] = cn;
  hs_t[idx] = (__bf16)(so * tanhf(cn));
}

// ---------------------------------------------------------------------------
extern "C" void kernel_launch(void* const* d_in, const int* in_sizes, int n_in,
                              void* d_out, int out_size, void* d_ws, size_t ws_size,
                              hipStream_t stream) {
  const float* z        = (const float*)d_in[0];
  const float* gru_Wih  = (const float*)d_in[1];
  const float* gru_Whh  = (const float*)d_in[2];
  const float* gru_bih  = (const float*)d_in[3];
  const float* gru_bhh  = (const float*)d_in[4];
  const float* lstm_Wih = (const float*)d_in[5];
  const float* lstm_Whh = (const float*)d_in[6];
  const float* lstm_bih = (const float*)d_in[7];
  const float* lstm_bhh = (const float*)d_in[8];
  const float* fc_W     = (const float*)d_in[9];
  const float* fc_b     = (const float*)d_in[10];
  // d_in[11] = sequence_length (device scalar); fixed at 512 for this module.

  char* wp = (char*)d_ws;
  auto alloc = [&](size_t bytes) -> char* {
    char* p = wp;
    wp += (bytes + 255) & ~(size_t)255;
    return p;
  };

  __bf16* Wih_g = (__bf16*)alloc((size_t)3 * DEC_L * DEC_L * 2);
  __bf16* Whh_g = (__bf16*)alloc((size_t)3 * DEC_L * DEC_L * 2);
  __bf16* Wih_l = (__bf16*)alloc((size_t)4 * DEC_L * DEC_L * 2);
  __bf16* Whh_l = (__bf16*)alloc((size_t)4 * DEC_L * DEC_L * 2);
  __bf16* Wfc   = (__bf16*)alloc((size_t)DEC_O * DEC_L * 2);
  __bf16* seq   = (__bf16*)alloc((size_t)DEC_T * DEC_BL * 2);  // GRU out (T,B,L)
  __bf16* hs    = (__bf16*)alloc((size_t)DEC_T * DEC_BL * 2);  // LSTM out (T,B,H)
  float*  h_gru = (float*)alloc((size_t)DEC_BL * 4);
  __bf16* h_gbf = (__bf16*)alloc((size_t)DEC_BL * 2);
  float*  c_lst = (float*)alloc((size_t)DEC_BL * 4);
  float*  gi_ws = (float*)alloc((size_t)3 * DEC_L * DEC_B * 4);
  float*  gh_ws = (float*)alloc((size_t)3 * DEC_L * DEC_B * 4);
  float*  gx_ws = (float*)alloc((size_t)4 * DEC_L * DEC_B * 4);
  float*  gl_ws = (float*)alloc((size_t)4 * DEC_L * DEC_B * 4);

  // ---- weight conversion fp32 -> bf16 (L2 resident thereafter) ----
  cvt_bf16_kernel<<<1024, 256, 0, stream>>>(gru_Wih,  Wih_g, 3 * DEC_L * DEC_L);
  cvt_bf16_kernel<<<1024, 256, 0, stream>>>(gru_Whh,  Whh_g, 3 * DEC_L * DEC_L);
  cvt_bf16_kernel<<<1024, 256, 0, stream>>>(lstm_Wih, Wih_l, 4 * DEC_L * DEC_L);
  cvt_bf16_kernel<<<1024, 256, 0, stream>>>(lstm_Whh, Whh_l, 4 * DEC_L * DEC_L);
  cvt_bf16_kernel<<<128,  256, 0, stream>>>(fc_W,     Wfc,   DEC_O * DEC_L);

  init_kernel<<<DEC_BL / 256, 256, 0, stream>>>(z, seq, h_gru, h_gbf, c_lst, DEC_BL);

  dim3 blk(256);
  dim3 g3(3 * DEC_L / 128, 1);   // 24 blocks
  dim3 g4(4 * DEC_L / 128, 1);   // 32 blocks

  // ---- autoregressive GRU: seq[t] for t = 1..T-1 ----
  for (int t = 1; t < DEC_T; ++t) {
    const __bf16* xprev = seq + (size_t)(t - 1) * DEC_BL;   // x input (== h for t>=2)
    wmma_gemm_nt<<<g3, blk, GEMM_LDS_BYTES, stream>>>(Wih_g, xprev, gi_ws,
                                                      DEC_L, DEC_B, DEC_B);
    wmma_gemm_nt<<<g3, blk, GEMM_LDS_BYTES, stream>>>(Whh_g, h_gbf, gh_ws,
                                                      DEC_L, DEC_B, DEC_B);
    gru_point_kernel<<<DEC_BL / 256, 256, 0, stream>>>(
        gi_ws, gh_ws, gru_bih, gru_bhh, h_gru,
        seq + (size_t)t * DEC_BL, h_gbf);
  }

  // ---- LSTM over seq, t = 0..T-1 (h0 = c0 = 0; seq[0] is the zero slab) ---
  for (int t = 0; t < DEC_T; ++t) {
    const __bf16* xt    = seq + (size_t)t * DEC_BL;
    const __bf16* hprev = (t == 0) ? seq : hs + (size_t)(t - 1) * DEC_BL;
    wmma_gemm_nt<<<g4, blk, GEMM_LDS_BYTES, stream>>>(Wih_l, xt,    gx_ws,
                                                      DEC_L, DEC_B, DEC_B);
    wmma_gemm_nt<<<g4, blk, GEMM_LDS_BYTES, stream>>>(Whh_l, hprev, gl_ws,
                                                      DEC_L, DEC_B, DEC_B);
    lstm_point_kernel<<<DEC_BL / 256, 256, 0, stream>>>(
        gx_ws, gl_ws, lstm_bih, lstm_bhh, c_lst, hs + (size_t)t * DEC_BL);
  }

  // ---- FC head: fused GEMM + bias + (B,T,O) scatter ----
  dim3 gf(1, DEC_T * DEC_B / 128);   // (1, 512)
  wmma_fc_kernel<<<gf, blk, 0, stream>>>(Wfc, hs, fc_b, (float*)d_out, DEC_L);
}
